// QuantizedLinear_45586782880462
// MI455X (gfx1250) — compile-verified, matrix operand order checked by reading, not run on hardware
//
#include <hip/hip_runtime.h>

typedef __attribute__((ext_vector_type(16))) _Float16 v16h;
typedef __attribute__((ext_vector_type(8)))  float    v8f;

#define TOKENS 8192
#define INF    4096
#define OUTF   11008

#define BM  128
#define BN  128
#define BK  32
#define PD  36                 // LDS tile pitch in dwords: rows 16B-aligned, banks spread
#define NK  (INF / BK)         // 128 K-steps

#define XTILE_DW (BM * PD)     // 4608 dwords per x tile buffer
#define WTILE_DW (BN * PD)     // 4608 dwords per w tile buffer
#define TILE_BYTES (XTILE_DW * 4u)   // 18432 B, same for both tiles -> one toggle

// ---------------------------------------------------------------------------
// Pass 1: per-token row sums of x (zero_point term).
// ---------------------------------------------------------------------------
__global__ __launch_bounds__(256) void rowsum_kernel(const float* __restrict__ x,
                                                     float* __restrict__ rs) {
    __shared__ float red[256];
    const int row = blockIdx.x;
    const float4* xr = (const float4*)(x + (size_t)row * INF);
    float s = 0.f;
#pragma unroll
    for (int i = 0; i < 4; ++i) {
        float4 v = xr[threadIdx.x + i * 256];
        s += v.x + v.y + v.z + v.w;
    }
    red[threadIdx.x] = s;
    __syncthreads();
    for (int off = 128; off > 0; off >>= 1) {
        if (threadIdx.x < off) red[threadIdx.x] += red[threadIdx.x + off];
        __syncthreads();
    }
    if (threadIdx.x == 0) rs[row] = red[0];
}

// ---------------------------------------------------------------------------
// Pass 2: WMMA GEMM with async global->LDS double buffering.
// out = scale * (x @ wq^T) + zp * rowsum + bias
// 256 threads = 8 waves (2x4); each wave computes 4x2 WMMA 16x16 tiles.
// ---------------------------------------------------------------------------
__global__ __launch_bounds__(256, 1) void qlin_wmma_kernel(
    const float* __restrict__ x,       // [TOKENS, INF] f32
    const int*   __restrict__ wq,      // [OUTF, INF] int8 in int32 container
    const float* __restrict__ scale_p, // [1]
    const float* __restrict__ zp_p,    // [1]
    const float* __restrict__ bias,    // [OUTF]
    const float* __restrict__ rowsum,  // [TOKENS]
    float*       __restrict__ out)     // [TOKENS, OUTF]
{
    // single static LDS block: [2 x raw f32 x-tile][2 x raw i32 w-tile]
    __shared__ float smem[2 * XTILE_DW + 2 * WTILE_DW];   // 72 KB
    float* Xs = smem;                                      // [2][XTILE_DW]
    int*   Wt = (int*)(smem + 2 * XTILE_DW);               // [2][WTILE_DW]

    // CRITICAL: the async copies (inline asm) are the only writers of smem and
    // they only receive integer LDS offsets. Escape smem's address into opaque
    // asm so the compiler must assume later "memory"-clobbering asm writes it;
    // otherwise every read of smem folds to undef (observed in round 2/3).
    {
        const void* esc = (const void*)smem;
        asm volatile("" :: "s"(esc) : "memory");
    }

    const int tid    = threadIdx.x;
    const int lane   = tid & 31;
    const int wave   = tid >> 5;
    const int wm     = wave >> 2;          // 0..1 -> 64-row strip
    const int wn     = wave & 3;           // 0..3 -> 32-col strip
    const int half16 = lane >> 4;          // 0 | 1
    const int l16    = lane & 15;

    const int m0 = blockIdx.y * BM;
    const int n0 = blockIdx.x * BN;

    // ---- per-thread async-copy slots: 1024 16-byte chunks per tile ----
    unsigned gx[4], gw[4], lxs[4], lws[4];
#pragma unroll
    for (int i = 0; i < 4; ++i) {
        const int idx = tid + i * 256;
        const int r   = idx >> 3;
        const int c   = (idx & 7) << 2;    // dword column
        gx[i] = ((unsigned)(m0 + r) * INF + c) * 4u;               // byte off vs x
        gw[i] = ((unsigned)(n0 + r) * INF + c) * 4u;               // byte off vs wq
        lxs[i] = (unsigned)((const char*)&Xs[r * PD + c] - (const char*)smem);
        lws[i] = (unsigned)((const char*)&Wt[r * PD + c] - (const char*)smem);
    }

    v8f acc[4][2] = {};

    // ---- prologue: async-fill buffer 0 ----
#pragma unroll
    for (int i = 0; i < 4; ++i) {
        asm volatile("global_load_async_to_lds_b128 %0, %1, %2"
                     :: "v"(lxs[i]), "v"(gx[i]), "s"(x) : "memory");
    }
#pragma unroll
    for (int i = 0; i < 4; ++i) {
        asm volatile("global_load_async_to_lds_b128 %0, %1, %2"
                     :: "v"(lws[i]), "v"(gw[i]), "s"(wq) : "memory");
    }
    asm volatile("s_wait_asynccnt 0" ::: "memory");
    __syncthreads();

    const int ko = half16 * 8;   // lanes 0-15: K {0..7,16..23}; lanes 16-31: +8

    unsigned bufOff = 0u;        // byte offset of the CURRENT buffer (0 or 18432)
    unsigned kb     = BK * 4u;   // global byte advance for tile kt+1

    // ---- main K loop: single body, dynamic double-buffer toggle ----
#pragma unroll 1
    for (int kt = 0; kt < NK; ++kt) {
        if (kt + 1 < NK) {
            const unsigned nb = bufOff ^ TILE_BYTES;      // next buffer
#pragma unroll
            for (int i = 0; i < 4; ++i) {
                unsigned l = lxs[i] + nb;
                unsigned g = gx[i] + kb;
                asm volatile("global_load_async_to_lds_b128 %0, %1, %2"
                             :: "v"(l), "v"(g), "s"(x) : "memory");
            }
#pragma unroll
            for (int i = 0; i < 4; ++i) {
                unsigned l = lws[i] + nb;
                unsigned g = gw[i] + kb;
                asm volatile("global_load_async_to_lds_b128 %0, %1, %2"
                             :: "v"(l), "v"(g), "s"(wq) : "memory");
            }
        }

        const float* xbase = (const float*)((const char*)smem + bufOff);
        const int*   wbase = (const int*)((const char*)smem + 2u * TILE_BYTES + bufOff);

        // B fragments (held across the WMMA block)
        v16h bf[2];
#pragma unroll
        for (int ni = 0; ni < 2; ++ni) {
            const int* p = wbase + (wn * 32 + ni * 16 + l16) * PD + ko;
            int4 d0 = *(const int4*)(p);
            int4 d1 = *(const int4*)(p + 4);
            int4 d2 = *(const int4*)(p + 16);
            int4 d3 = *(const int4*)(p + 20);
            v16h b;
            b[0]=(_Float16)d0.x; b[1]=(_Float16)d0.y; b[2]=(_Float16)d0.z; b[3]=(_Float16)d0.w;
            b[4]=(_Float16)d1.x; b[5]=(_Float16)d1.y; b[6]=(_Float16)d1.z; b[7]=(_Float16)d1.w;
            b[8]=(_Float16)d2.x; b[9]=(_Float16)d2.y; b[10]=(_Float16)d2.z; b[11]=(_Float16)d2.w;
            b[12]=(_Float16)d3.x; b[13]=(_Float16)d3.y; b[14]=(_Float16)d3.z; b[15]=(_Float16)d3.w;
            bf[ni] = b;
        }

        // All four A fragments materialized before the WMMA block, so nothing
        // overwrites a WMMA source until far after its last read (no WAR nops).
        v16h af[4];
#pragma unroll
        for (int mi = 0; mi < 4; ++mi) {
            const float* p = xbase + (wm * 64 + mi * 16 + l16) * PD + ko;
            float4 c0 = *(const float4*)(p);
            float4 c1 = *(const float4*)(p + 4);
            float4 c2 = *(const float4*)(p + 16);
            float4 c3 = *(const float4*)(p + 20);
            v16h a;
            a[0]=(_Float16)c0.x; a[1]=(_Float16)c0.y; a[2]=(_Float16)c0.z; a[3]=(_Float16)c0.w;
            a[4]=(_Float16)c1.x; a[5]=(_Float16)c1.y; a[6]=(_Float16)c1.z; a[7]=(_Float16)c1.w;
            a[8]=(_Float16)c2.x; a[9]=(_Float16)c2.y; a[10]=(_Float16)c2.z; a[11]=(_Float16)c2.w;
            a[12]=(_Float16)c3.x; a[13]=(_Float16)c3.y; a[14]=(_Float16)c3.z; a[15]=(_Float16)c3.w;
            af[mi] = a;
        }

#pragma unroll
        for (int mi = 0; mi < 4; ++mi) {
            acc[mi][0] = __builtin_amdgcn_wmma_f32_16x16x32_f16(
                false, af[mi], false, bf[0], (short)0, acc[mi][0], false, false);
            acc[mi][1] = __builtin_amdgcn_wmma_f32_16x16x32_f16(
                false, af[mi], false, bf[1], (short)0, acc[mi][1], false, false);
        }

        // this wave's async copies done, then block-wide visibility
        asm volatile("s_wait_asynccnt 0" ::: "memory");
        __syncthreads();

        bufOff ^= TILE_BYTES;
        kb     += BK * 4u;
    }

    // ---- epilogue: out = scale*acc + zp*rowsum[m] + bias[n] ----
    const float s = scale_p[0];
    const float z = zp_p[0];
#pragma unroll
    for (int mi = 0; mi < 4; ++mi) {
        const int mbase = m0 + wm * 64 + mi * 16 + half16 * 8;  // lanes>=16 -> M+8
#pragma unroll
        for (int ni = 0; ni < 2; ++ni) {
            const int n = n0 + wn * 32 + ni * 16 + l16;         // N = lane%16
            const float b = bias[n];
#pragma unroll
            for (int r = 0; r < 8; ++r) {
                const int m = mbase + r;                        // VGPR r -> M%8
                out[(size_t)m * OUTF + n] = s * acc[mi][ni][r] + z * rowsum[m] + b;
            }
        }
    }
}

// ---------------------------------------------------------------------------
extern "C" void kernel_launch(void* const* d_in, const int* in_sizes, int n_in,
                              void* d_out, int out_size, void* d_ws, size_t ws_size,
                              hipStream_t stream) {
    (void)in_sizes; (void)n_in; (void)out_size; (void)ws_size;
    const float* x     = (const float*)d_in[0];
    const int*   wq    = (const int*)  d_in[1];
    const float* scale = (const float*)d_in[2];
    const float* zp    = (const float*)d_in[3];
    const float* bias  = (const float*)d_in[4];
    float*       out   = (float*)d_out;
    float*       rs    = (float*)d_ws;          // TOKENS floats = 32 KB scratch

    rowsum_kernel<<<TOKENS, 256, 0, stream>>>(x, rs);

    dim3 grid(OUTF / BN, TOKENS / BM);          // 86 x 64 blocks
    qlin_wmma_kernel<<<grid, 256, 0, stream>>>(x, wq, scale, zp, bias, rs, out);
}
